// DecoderRNN_73117523247478
// MI455X (gfx1250) — matrix operand order repaired and checked by reading, not compiled, not run on hardware
//
#include <hip/hip_runtime.h>
#include <hip/hip_bf16.h>

// ---------------------------------------------------------------------------
// DecoderRNN (greedy-feedback LSTM + CE loss) for MI455X / gfx1250
// B=128, S=256, D=H=1024, T=20, START_IDX=19
//
// Per-step gates GEMM: v_wmma_f32_16x16x32_bf16 (wave32), with the B (weight)
// tile staged through LDS via global_load_async_to_lds_b128 (ASYNCcnt),
// double-buffered, and A fragments software-pipelined in registers.
// B fragments for all 4 N-tiles are loaded before the 4 WMMAs so the ds_load
// latency is paid once per chunk, not once per WMMA.
// ---------------------------------------------------------------------------

#define BATCH 128
#define SEQ   256
#define DIM   1024
#define HID   1024
#define TAGS  20
#define START_IDX 19
#define G4H   4096          // 4*HID
#define KTOT  2048          // DIM + HID (concat GEMM K)
#define KCH   32            // K chunk per WMMA step
#define NCHUNKS (KTOT / KCH)

typedef __attribute__((ext_vector_type(16))) __bf16 v16bf;
typedef __attribute__((ext_vector_type(8)))  __bf16 bf16x8;
typedef __attribute__((ext_vector_type(8)))  float  v8f;

// ---------------------------------------------------------------------------
// Setup kernels
// ---------------------------------------------------------------------------
__global__ __launch_bounds__(256) void cvt_x_kernel(const float* __restrict__ x,
                                                    __hip_bfloat16* __restrict__ xb,
                                                    long n) {
  long i = (long)blockIdx.x * 256 + threadIdx.x;
  if (i < n) xb[i] = __float2bfloat16(x[i]);
}

// Wcat[n][k] : k<1024 -> W_ih[n][k] (row stride 1044), else W_hh[n][k-1024]
__global__ __launch_bounds__(256) void build_wcat_kernel(const float* __restrict__ W_ih,
                                                         const float* __restrict__ W_hh,
                                                         __hip_bfloat16* __restrict__ wcat) {
  long i = (long)blockIdx.x * 256 + threadIdx.x;
  if (i >= (long)G4H * KTOT) return;
  int n = (int)(i >> 11);
  int k = (int)(i & (KTOT - 1));
  float v = (k < DIM) ? W_ih[(size_t)n * (DIM + TAGS) + k]
                      : W_hh[(size_t)n * HID + (k - DIM)];
  wcat[i] = __float2bfloat16(v);
}

__global__ __launch_bounds__(256) void init_state_kernel(float* __restrict__ c,
                                                         float* __restrict__ h_f32,
                                                         __hip_bfloat16* __restrict__ h_bf,
                                                         int* __restrict__ prev_tag,
                                                         float* __restrict__ loss) {
  int i = blockIdx.x * 256 + threadIdx.x;
  if (i < BATCH * HID) {
    c[i] = 0.f;
    h_f32[i] = 0.f;
    h_bf[i] = __float2bfloat16(0.f);
  }
  if (i < BATCH) prev_tag[i] = START_IDX;
  if (i == 0) loss[0] = 0.f;
}

// ---------------------------------------------------------------------------
// Per-step GEMM: gates[128,4096] = [x_t | h] (bf16) @ Wcat^T (bf16), f32 acc.
// Block = 8 waves = 8 M-tiles (all 128 rows) x one 64-column N group.
// Grid = 64 blocks (N groups). K = 2048 in chunks of 32.
//
// B chunk (64 cols x 32 k, 4 KB) is async-copied to LDS once per chunk by the
// whole block (one global_load_async_to_lds_b128 per thread), double-buffered;
// waves read fragments via ds_load_b128. A fragments pipelined in registers.
//
// bf16 A fragment per-lane layout (this code's contiguous k order, 16-bit A 16x32):
//   lane = half*16 + m;  a[0..15]=K[kk+16*half .. +15]
// bf16 B fragment (32x16): lane = half*16 + n; b[0..15]=K[kk+16*half .. +15]
// C/D f32 16x16: VGPR r, lane -> (m = r + 8*half, n = lane&15)
// ---------------------------------------------------------------------------
__global__ __launch_bounds__(256)
void gemm_step_kernel(const __hip_bfloat16* __restrict__ Xb,   // [B,S,D] bf16
                      const __hip_bfloat16* __restrict__ Hb,   // [B,H] bf16
                      const __hip_bfloat16* __restrict__ Wcat, // [4096,2048] bf16
                      float* __restrict__ gates,               // [B,4096] f32
                      int t) {
  __shared__ __align__(16) __hip_bfloat16 lbuf[2][64 * KCH];   // 2 x 4 KB

  const int tid   = threadIdx.x;
  const int lane  = tid & 31;
  const int wave  = tid >> 5;          // wave = M tile (0..7)
  const int half  = lane >> 4;
  const int nlane = lane & 15;
  const int ncol0 = blockIdx.x * 64;   // 64 N groups

  // async-copy addressing: this thread moves one 16B chunk per k-step
  const int ccol = tid >> 2;           // 0..63 : local column
  const int cq   = tid & 3;            // 0..3  : 8-element quarter of 32 k
  const __hip_bfloat16* wsrc = Wcat + (size_t)(ncol0 + ccol) * KTOT + cq * 8;

  const int mrow = wave * 16 + nlane;  // A row carried by this lane
  const __hip_bfloat16* xrow = Xb + ((size_t)mrow * SEQ + t) * DIM;
  const __hip_bfloat16* hrow = Hb + (size_t)mrow * HID;

  v8f acc0 = {}, acc1 = {}, acc2 = {}, acc3 = {};

  // --- prologue: async-prefetch B chunk 0, load A chunk 0 ---
  {
    unsigned lds_off = (unsigned)(size_t)&lbuf[0][ccol * KCH + cq * 8];
    unsigned long long ga = (unsigned long long)(size_t)wsrc;
    asm volatile("global_load_async_to_lds_b128 %0, %1, off"
                 :: "v"(lds_off), "v"(ga) : "memory");
  }
  bf16x8 alo = *(const bf16x8*)(const void*)(xrow + 16 * half);
  bf16x8 ahi = *(const bf16x8*)(const void*)(xrow + 16 * half + 8);
  asm volatile("s_wait_asynccnt 0x0" ::: "memory");
  __syncthreads();

  for (int i = 0; i < NCHUNKS; ++i) {
    const int cur = i & 1;
    const int kk  = i * KCH;

    // async-prefetch next B chunk into the other buffer
    if (i + 1 < NCHUNKS) {
      unsigned lds_off = (unsigned)(size_t)&lbuf[cur ^ 1][ccol * KCH + cq * 8];
      unsigned long long ga = (unsigned long long)(size_t)(wsrc + kk + KCH);
      asm volatile("global_load_async_to_lds_b128 %0, %1, off"
                   :: "v"(lds_off), "v"(ga) : "memory");
    }
    // prefetch next A fragment (registers)
    bf16x8 nlo = alo, nhi = ahi;
    if (i + 1 < NCHUNKS) {
      const int kn = kk + KCH;
      const __hip_bfloat16* s =
          (kn < DIM ? xrow + kn : hrow + (kn - DIM)) + 16 * half;
      nlo = *(const bf16x8*)(const void*)(s);
      nhi = *(const bf16x8*)(const void*)(s + 8);
    }

    v16bf a;
#pragma unroll
    for (int e = 0; e < 8; ++e) { a[e] = alo[e]; a[8 + e] = ahi[e]; }

    // Load ALL four B fragments first (distinct registers), then issue the
    // four WMMAs back-to-back: one dscnt stall per chunk instead of four.
    v16bf b[4];
#pragma unroll
    for (int nn = 0; nn < 4; ++nn) {
      const __hip_bfloat16* lb =
          &lbuf[cur][(nn * 16 + nlane) * KCH + 16 * half];
      bf16x8 blo = *(const bf16x8*)(const void*)(lb);
      bf16x8 bhi = *(const bf16x8*)(const void*)(lb + 8);
#pragma unroll
      for (int e = 0; e < 8; ++e) { b[nn][e] = blo[e]; b[nn][8 + e] = bhi[e]; }
    }

    acc0 = __builtin_amdgcn_wmma_f32_16x16x32_bf16(
        false, a, false, b[0], (short)0, acc0, false, false);
    acc1 = __builtin_amdgcn_wmma_f32_16x16x32_bf16(
        false, a, false, b[1], (short)0, acc1, false, false);
    acc2 = __builtin_amdgcn_wmma_f32_16x16x32_bf16(
        false, a, false, b[2], (short)0, acc2, false, false);
    acc3 = __builtin_amdgcn_wmma_f32_16x16x32_bf16(
        false, a, false, b[3], (short)0, acc3, false, false);

    alo = nlo; ahi = nhi;
    // next B chunk arrived; all waves done reading current buffer
    asm volatile("s_wait_asynccnt 0x0" ::: "memory");
    __syncthreads();
  }

#pragma unroll
  for (int nn = 0; nn < 4; ++nn) {
    const v8f acc = (nn == 0) ? acc0 : (nn == 1) ? acc1 : (nn == 2) ? acc2 : acc3;
    const int n = ncol0 + nn * 16 + nlane;
#pragma unroll
    for (int r = 0; r < 8; ++r) {
      const int m = wave * 16 + r + 8 * half;
      gates[(size_t)m * G4H + n] = acc[r];
    }
  }
}

// ---------------------------------------------------------------------------
// Elementwise LSTM cell: add biases + one-hot tag column, gate nonlinearities.
// Gate order (PyTorch): i, f, g, o. 128*1024 threads.
// ---------------------------------------------------------------------------
__global__ __launch_bounds__(256)
void lstm_cell_kernel(const float* __restrict__ gates,
                      const float* __restrict__ b_ih,
                      const float* __restrict__ b_hh,
                      const float* __restrict__ W_ih,   // [4096, 1044] f32
                      const int* __restrict__ prev_tag,
                      float* __restrict__ c,
                      float* __restrict__ h_f32,
                      __hip_bfloat16* __restrict__ h_bf) {
  const int idx = blockIdx.x * 256 + threadIdx.x;    // 0 .. 128*1024-1
  const int b = idx >> 10;
  const int j = idx & 1023;
  const int tag = prev_tag[b];
  const size_t base = (size_t)b * G4H;
  const int RW = DIM + TAGS;   // 1044

  float gi = gates[base + j]
           + b_ih[j] + b_hh[j] + W_ih[(size_t)j * RW + DIM + tag];
  float gf = gates[base + 1024 + j]
           + b_ih[1024 + j] + b_hh[1024 + j] + W_ih[(size_t)(1024 + j) * RW + DIM + tag];
  float gg = gates[base + 2048 + j]
           + b_ih[2048 + j] + b_hh[2048 + j] + W_ih[(size_t)(2048 + j) * RW + DIM + tag];
  float go = gates[base + 3072 + j]
           + b_ih[3072 + j] + b_hh[3072 + j] + W_ih[(size_t)(3072 + j) * RW + DIM + tag];

  float i_ = 1.f / (1.f + __expf(-gi));
  float f_ = 1.f / (1.f + __expf(-gf));
  float g_ = tanhf(gg);
  float o_ = 1.f / (1.f + __expf(-go));

  float cn = f_ * c[idx] + i_ * g_;
  float hn = o_ * tanhf(cn);
  c[idx] = cn;
  h_f32[idx] = hn;
  h_bf[idx] = __float2bfloat16(hn);
}

// ---------------------------------------------------------------------------
// Logits + argmax feedback + CE loss. One block (256 thr = 8 wave32) per row.
// ---------------------------------------------------------------------------
__global__ __launch_bounds__(256)
void logits_step_kernel(const float* __restrict__ h_f32,
                        const float* __restrict__ W_out,  // [20,1024]
                        const float* __restrict__ b_out,
                        const int* __restrict__ tags,     // [B,S]
                        const int* __restrict__ mask,     // [B,S]
                        int t,
                        int* __restrict__ prev_tag,
                        float* __restrict__ loss) {
  const int b = blockIdx.x;
  const int tid = threadIdx.x;
  const float* hrow = h_f32 + (size_t)b * HID;

  float acc[TAGS];
#pragma unroll
  for (int k = 0; k < TAGS; ++k) acc[k] = 0.f;

  for (int d = tid; d < HID; d += 256) {
    const float hv = hrow[d];
#pragma unroll
    for (int k = 0; k < TAGS; ++k) acc[k] += hv * W_out[(size_t)k * HID + d];
  }

  // wave32 tree reduce
#pragma unroll
  for (int off = 16; off > 0; off >>= 1) {
#pragma unroll
    for (int k = 0; k < TAGS; ++k) acc[k] += __shfl_down(acc[k], off, 32);
  }

  __shared__ float red[8][TAGS];
  if ((tid & 31) == 0) {
#pragma unroll
    for (int k = 0; k < TAGS; ++k) red[tid >> 5][k] = acc[k];
  }
  __syncthreads();

  if (tid == 0) {
    float logit[TAGS];
#pragma unroll
    for (int k = 0; k < TAGS; ++k) {
      float s = 0.f;
#pragma unroll
      for (int w = 0; w < 8; ++w) s += red[w][k];
      logit[k] = s + b_out[k];
    }
    // first-occurrence argmax (matches jnp.argmax)
    int am = 0;
    float mx = logit[0];
#pragma unroll
    for (int k = 1; k < TAGS; ++k)
      if (logit[k] > mx) { mx = logit[k]; am = k; }
    prev_tag[b] = am;

    // log-softmax CE for this (b, t)
    float s = 0.f;
#pragma unroll
    for (int k = 0; k < TAGS; ++k) s += __expf(logit[k] - mx);
    const float lse = logf(s) + mx;

    const int y = tags[(size_t)b * SEQ + t];
    const int mk = mask[(size_t)b * SEQ + t];
    if (mk != 0) atomicAdd(loss, lse - logit[y]);
  }
}

// ---------------------------------------------------------------------------
// Host side
// ---------------------------------------------------------------------------
extern "C" void kernel_launch(void* const* d_in, const int* in_sizes, int n_in,
                              void* d_out, int out_size, void* d_ws, size_t ws_size,
                              hipStream_t stream) {
  (void)in_sizes; (void)n_in; (void)out_size; (void)ws_size;

  const float* x     = (const float*)d_in[0];   // [B,S,D]
  const int*   tags  = (const int*)  d_in[1];   // [B,S]
  const int*   mask  = (const int*)  d_in[2];   // [B,S]
  const float* W_ih  = (const float*)d_in[3];   // [4096,1044]
  const float* W_hh  = (const float*)d_in[4];   // [4096,1024]
  const float* b_ih  = (const float*)d_in[5];
  const float* b_hh  = (const float*)d_in[6];
  const float* W_out = (const float*)d_in[7];   // [20,1024]
  const float* b_out = (const float*)d_in[8];
  float* out = (float*)d_out;                   // scalar loss

  // workspace layout (all offsets 512B-aligned); ~84 MB total
  char* ws = (char*)d_ws;
  const size_t XB_BYTES   = (size_t)BATCH * SEQ * DIM * 2;  // 64 MB
  const size_t WC_BYTES   = (size_t)G4H * KTOT * 2;         // 16 MB
  const size_t GT_BYTES   = (size_t)BATCH * G4H * 4;        // 2 MB
  const size_t HF_BYTES   = (size_t)BATCH * HID * 4;
  const size_t C_BYTES    = (size_t)BATCH * HID * 4;
  const size_t HB_BYTES   = (size_t)BATCH * HID * 2;

  __hip_bfloat16* Xb    = (__hip_bfloat16*)(ws);
  __hip_bfloat16* Wcat  = (__hip_bfloat16*)(ws + XB_BYTES);
  float*          gates = (float*)(ws + XB_BYTES + WC_BYTES);
  float*          h_f32 = (float*)(ws + XB_BYTES + WC_BYTES + GT_BYTES);
  float*          c     = (float*)(ws + XB_BYTES + WC_BYTES + GT_BYTES + HF_BYTES);
  __hip_bfloat16* h_bf  = (__hip_bfloat16*)(ws + XB_BYTES + WC_BYTES + GT_BYTES + HF_BYTES + C_BYTES);
  int*            ptag  = (int*)(ws + XB_BYTES + WC_BYTES + GT_BYTES + HF_BYTES + C_BYTES + HB_BYTES);

  const long nx = (long)BATCH * SEQ * DIM;
  cvt_x_kernel<<<(nx + 255) / 256, 256, 0, stream>>>(x, Xb, nx);

  const long nw = (long)G4H * KTOT;
  build_wcat_kernel<<<(int)((nw + 255) / 256), 256, 0, stream>>>(W_ih, W_hh, Wcat);

  init_state_kernel<<<(BATCH * HID + 255) / 256, 256, 0, stream>>>(c, h_f32, h_bf, ptag, out);

  for (int t = 0; t < SEQ; ++t) {
    gemm_step_kernel<<<64, 256, 0, stream>>>(Xb, h_bf, Wcat, gates, t);
    lstm_cell_kernel<<<(BATCH * HID) / 256, 256, 0, stream>>>(gates, b_ih, b_hh, W_ih,
                                                              ptag, c, h_f32, h_bf);
    logits_step_kernel<<<BATCH, 256, 0, stream>>>(h_f32, W_out, b_out, tags, mask, t,
                                                  ptag, out);
  }
}
